// GAT_example_26336739459200
// MI455X (gfx1250) — matrix operand order repaired and checked by reading, not compiled, not run on hardware
//
#include <hip/hip_runtime.h>
#include <hip/hip_bf16.h>
#include <math.h>

typedef __attribute__((ext_vector_type(16))) _Float16 v16h;
typedef __attribute__((ext_vector_type(8)))  _Float16 v8h;
typedef __attribute__((ext_vector_type(8)))  float    v8f;

#define N_NODES 50000
#define N_EDGES 800000
#define E_TOT   850000          // edges + self loops
#define F_IN    128
#define HID     32
#define HEADS   8
#define C1      256             // HEADS*HID
#define N_CLS   2
#define NEG_SLOPE 0.2f

// ---------------- workspace layout (float offsets) ----------------
#define OFF_H1    0u                         // [N,256] f32 (reused as z after act)
#define OFF_OUT1  (OFF_H1   + N_NODES*C1)    // [N,256] f32
#define OFF_A1S   (OFF_OUT1 + N_NODES*C1)    // [N,8]
#define OFF_A1D   (OFF_A1S  + N_NODES*HEADS) // [N,8]
#define OFF_M1    (OFF_A1D  + N_NODES*HEADS) // [N,8]
#define OFF_S1    (OFF_M1   + N_NODES*HEADS) // [N,8]
#define OFF_H2    (OFF_S1   + N_NODES*HEADS) // [N,2]
#define OFF_A2S   (OFF_H2   + N_NODES*N_CLS) // [N]
#define OFF_A2D   (OFF_A2S  + N_NODES)       // [N]
#define OFF_M2    (OFF_A2D  + N_NODES)       // [N]
#define OFF_S2    (OFF_M2   + N_NODES)       // [N]
#define OFF_OUT2  (OFF_S2   + N_NODES)       // [N,2]
#define OFF_XH    (OFF_OUT2 + N_NODES*N_CLS) // [N,128] f16 -> N*64 float slots
#define OFF_W1T   (OFF_XH   + N_NODES*(F_IN/2)) // [256,128] f16 transposed -> 16384 float slots

static __device__ __forceinline__ float lrelu(float x) {
    return x > 0.f ? x : NEG_SLOPE * x;
}

// float atomic max via signed/unsigned integer views (valid for non-NaN data)
static __device__ __forceinline__ void atomicMaxF32(float* addr, float v) {
    if (v >= 0.f) atomicMax((int*)addr, __float_as_int(v));
    else          atomicMin((unsigned int*)addr, __float_as_uint(v));
}

// ---------------- init / convert kernels ----------------
__global__ void k_fill(float* p, unsigned n, float v) {
    unsigned i = blockIdx.x * blockDim.x + threadIdx.x;
    if (i < n) p[i] = v;
}

__global__ void k_cvt_x(const float* __restrict__ x, _Float16* __restrict__ xh) {
    unsigned i = blockIdx.x * blockDim.x + threadIdx.x;
    unsigned base = i * 4;
    if (base >= (unsigned)N_NODES * F_IN) return;
    float4 v = *(const float4*)(x + base);
    v8h dummy; (void)dummy;
    xh[base + 0] = (_Float16)v.x;
    xh[base + 1] = (_Float16)v.y;
    xh[base + 2] = (_Float16)v.z;
    xh[base + 3] = (_Float16)v.w;
}

// W1 [128,256] f32 row-major  ->  w1t [256,128] f16 (column-major of W1)
__global__ void k_cvt_w1t(const float* __restrict__ W1, _Float16* __restrict__ w1t) {
    unsigned i = blockIdx.x * blockDim.x + threadIdx.x;
    if (i >= (unsigned)F_IN * C1) return;
    unsigned k = i / C1;
    unsigned c = i - k * C1;
    w1t[c * F_IN + k] = (_Float16)W1[i];
}

// ---------------- layer-1 GEMM: h1 = xh @ W1 (f16 in, f32 accum) ----------------
// one 16x16 output tile per wave; K = 128 in 4 steps of 32; all operands pre-f16
__global__ __launch_bounds__(128) void k_gemm1_wmma(const _Float16* __restrict__ xh,
                                                    const _Float16* __restrict__ w1t,
                                                    float* __restrict__ h1) {
    const int warp = threadIdx.x >> 5;
    const int lane = threadIdx.x & 31;
    const int t    = blockIdx.x * 4 + warp;          // tile id (wave-uniform)
    if (t >= (N_NODES / 16) * (C1 / 16)) return;     // drops whole waves only
    const int mt   = t >> 4;                         // 0..3124
    const int nt   = t & 15;                         // 0..15
    const int lsub = lane & 15;
    const int half = lane >> 4;
    const int row  = mt * 16 + lsub;                 // A row for this lane
    const int col  = nt * 16 + lsub;                 // B/C col for this lane

    const _Float16* arow = xh  + (unsigned)row * F_IN;
    const _Float16* bcol = w1t + (unsigned)col * F_IN;

    v8f acc = {};
    #pragma unroll
    for (int kb = 0; kb < F_IN; kb += 32) {
        // A 16x32 f16 layout: e<8 -> K = 8*half+e ; e>=8 -> K = 16+8*half+(e-8)
        v8h a_lo = *(const v8h*)(arow + kb + 8 * half);
        v8h a_hi = *(const v8h*)(arow + kb + 16 + 8 * half);
        v16h av = __builtin_shufflevector(a_lo, a_hi,
                     0,1,2,3,4,5,6,7,8,9,10,11,12,13,14,15);
        // B 32x16 f16 layout: e<8 -> K = 8*half+e ; e>=8 -> K = 16+8*half+(e-8)
        v8h b_lo = *(const v8h*)(bcol + kb + 8 * half);
        v8h b_hi = *(const v8h*)(bcol + kb + 16 + 8 * half);
        v16h bv = __builtin_shufflevector(b_lo, b_hi,
                     0,1,2,3,4,5,6,7,8,9,10,11,12,13,14,15);
        acc = __builtin_amdgcn_wmma_f32_16x16x32_f16(false, av, false, bv,
                                                     (short)0, acc, false, false);
    }
    // C/D: VGPR r holds row r + 8*half, col = lane%16
    #pragma unroll
    for (int r = 0; r < 8; ++r) {
        h1[(unsigned)(mt * 16 + r + 8 * half) * C1 + col] = acc[r];
    }
}

// ---------------- per-node attention scalars, layer 1 ----------------
__global__ void k_alpha1(const float* __restrict__ h1,
                         const float* __restrict__ a1_src,
                         const float* __restrict__ a1_dst,
                         float* __restrict__ a1s, float* __restrict__ a1d) {
    unsigned i = blockIdx.x * blockDim.x + threadIdx.x;   // (node, head)
    if (i >= (unsigned)N_NODES * HEADS) return;
    unsigned n  = i >> 3;
    unsigned hd = i & 7;
    const float* hp = h1 + n * C1 + hd * HID;
    const float* ws = a1_src + hd * HID;
    const float* wd = a1_dst + hd * HID;
    float s = 0.f, d = 0.f;
    #pragma unroll
    for (int c = 0; c < HID; c += 4) {
        float4 hv = *(const float4*)(hp + c);
        float4 sv = *(const float4*)(ws + c);
        float4 dv = *(const float4*)(wd + c);
        s += hv.x*sv.x + hv.y*sv.y + hv.z*sv.z + hv.w*sv.w;
        d += hv.x*dv.x + hv.y*dv.y + hv.z*dv.z + hv.w*dv.w;
    }
    a1s[i] = s;
    a1d[i] = d;
}

static __device__ __forceinline__ void edge_nodes(const int* ei, unsigned e,
                                                  int& src, int& dst) {
    if (e < N_EDGES) { src = ei[e]; dst = ei[N_EDGES + e]; }
    else             { src = dst = (int)(e - N_EDGES); }   // self loop
}

// ---------------- layer-1 edge passes ----------------
__global__ void k_edge_max1(const int* __restrict__ ei,
                            const float* __restrict__ a1s,
                            const float* __restrict__ a1d,
                            float* __restrict__ m1) {
    unsigned e = blockIdx.x * blockDim.x + threadIdx.x;
    if (e >= E_TOT) return;
    int src, dst; edge_nodes(ei, e, src, dst);
    #pragma unroll
    for (int hd = 0; hd < HEADS; ++hd) {
        float lg = lrelu(a1s[src * HEADS + hd] + a1d[dst * HEADS + hd]);
        atomicMaxF32(&m1[dst * HEADS + hd], lg);
    }
}

__global__ void k_edge_sum1(const int* __restrict__ ei,
                            const float* __restrict__ a1s,
                            const float* __restrict__ a1d,
                            const float* __restrict__ m1,
                            float* __restrict__ s1) {
    unsigned e = blockIdx.x * blockDim.x + threadIdx.x;
    if (e >= E_TOT) return;
    int src, dst; edge_nodes(ei, e, src, dst);
    #pragma unroll
    for (int hd = 0; hd < HEADS; ++hd) {
        float lg = lrelu(a1s[src * HEADS + hd] + a1d[dst * HEADS + hd]);
        float ex = __expf(lg - m1[dst * HEADS + hd]);
        atomicAdd(&s1[dst * HEADS + hd], ex);
    }
}

// one thread per (edge, head): gather 32 channels of h1[src], scatter-add to out1[dst]
__global__ void k_edge_msg1(const int* __restrict__ ei,
                            const float* __restrict__ a1s,
                            const float* __restrict__ a1d,
                            const float* __restrict__ m1,
                            const float* __restrict__ s1,
                            const float* __restrict__ h1,
                            float* __restrict__ out1) {
    unsigned i = blockIdx.x * blockDim.x + threadIdx.x;
    if (i >= (unsigned)E_TOT * HEADS) return;
    unsigned e  = i >> 3;
    unsigned hd = i & 7;
    int src, dst; edge_nodes(ei, e, src, dst);
    float lg = lrelu(a1s[src * HEADS + hd] + a1d[dst * HEADS + hd]);
    float w  = __expf(lg - m1[dst * HEADS + hd]) / (s1[dst * HEADS + hd] + 1e-16f);
    const float* hp = h1 + (unsigned)src * C1 + hd * HID;
    float* op       = out1 + (unsigned)dst * C1 + hd * HID;
    #pragma unroll
    for (int c = 0; c < HID; c += 4) {
        float4 hv = *(const float4*)(hp + c);
        atomicAdd(op + c + 0, hv.x * w);
        atomicAdd(op + c + 1, hv.y * w);
        atomicAdd(op + c + 2, hv.z * w);
        atomicAdd(op + c + 3, hv.w * w);
    }
}

// ---------------- bias + ELU, writes into z (reuses h1 region) ----------------
__global__ void k_act1(const float* __restrict__ out1,
                       const float* __restrict__ b1,
                       float* __restrict__ z) {
    unsigned i = blockIdx.x * blockDim.x + threadIdx.x;
    if (i >= (unsigned)N_NODES * C1) return;
    float v = out1[i] + b1[i & (C1 - 1)];
    z[i] = v > 0.f ? v : (__expf(v) - 1.f);
}

// ---------------- layer-2 features + attention scalars ----------------
__global__ void k_feat2(const float* __restrict__ z,
                        const float* __restrict__ W2,
                        const float* __restrict__ a2_src,
                        const float* __restrict__ a2_dst,
                        float* __restrict__ h2,
                        float* __restrict__ a2s, float* __restrict__ a2d) {
    unsigned n = blockIdx.x * blockDim.x + threadIdx.x;
    if (n >= N_NODES) return;
    const float* zp = z + n * C1;
    float acc0 = 0.f, acc1 = 0.f;
    #pragma unroll 4
    for (int j = 0; j < C1; j += 4) {
        float4 zv = *(const float4*)(zp + j);
        acc0 += zv.x * W2[(j+0)*N_CLS] + zv.y * W2[(j+1)*N_CLS]
              + zv.z * W2[(j+2)*N_CLS] + zv.w * W2[(j+3)*N_CLS];
        acc1 += zv.x * W2[(j+0)*N_CLS+1] + zv.y * W2[(j+1)*N_CLS+1]
              + zv.z * W2[(j+2)*N_CLS+1] + zv.w * W2[(j+3)*N_CLS+1];
    }
    h2[n * N_CLS]     = acc0;
    h2[n * N_CLS + 1] = acc1;
    a2s[n] = acc0 * a2_src[0] + acc1 * a2_src[1];
    a2d[n] = acc0 * a2_dst[0] + acc1 * a2_dst[1];
}

// ---------------- layer-2 edge passes (1 head, 2 channels) ----------------
__global__ void k_edge_max2(const int* __restrict__ ei,
                            const float* __restrict__ a2s,
                            const float* __restrict__ a2d,
                            float* __restrict__ m2) {
    unsigned e = blockIdx.x * blockDim.x + threadIdx.x;
    if (e >= E_TOT) return;
    int src, dst; edge_nodes(ei, e, src, dst);
    atomicMaxF32(&m2[dst], lrelu(a2s[src] + a2d[dst]));
}

__global__ void k_edge_sum2(const int* __restrict__ ei,
                            const float* __restrict__ a2s,
                            const float* __restrict__ a2d,
                            const float* __restrict__ m2,
                            float* __restrict__ s2) {
    unsigned e = blockIdx.x * blockDim.x + threadIdx.x;
    if (e >= E_TOT) return;
    int src, dst; edge_nodes(ei, e, src, dst);
    atomicAdd(&s2[dst], __expf(lrelu(a2s[src] + a2d[dst]) - m2[dst]));
}

__global__ void k_edge_msg2(const int* __restrict__ ei,
                            const float* __restrict__ a2s,
                            const float* __restrict__ a2d,
                            const float* __restrict__ m2,
                            const float* __restrict__ s2,
                            const float* __restrict__ h2,
                            float* __restrict__ out2) {
    unsigned e = blockIdx.x * blockDim.x + threadIdx.x;
    if (e >= E_TOT) return;
    int src, dst; edge_nodes(ei, e, src, dst);
    float w = __expf(lrelu(a2s[src] + a2d[dst]) - m2[dst]) / (s2[dst] + 1e-16f);
    atomicAdd(&out2[dst * N_CLS],     h2[src * N_CLS]     * w);
    atomicAdd(&out2[dst * N_CLS + 1], h2[src * N_CLS + 1] * w);
}

// ---------------- final: out = out2 + b2 (mean over 1 head == identity) ----------
__global__ void k_final(const float* __restrict__ out2,
                        const float* __restrict__ b2,
                        float* __restrict__ out) {
    unsigned i = blockIdx.x * blockDim.x + threadIdx.x;
    if (i >= (unsigned)N_NODES * N_CLS) return;
    out[i] = out2[i] + b2[i & 1];
}

static inline unsigned cdivu(unsigned a, unsigned b) { return (a + b - 1) / b; }

extern "C" void kernel_launch(void* const* d_in, const int* in_sizes, int n_in,
                              void* d_out, int out_size, void* d_ws, size_t ws_size,
                              hipStream_t stream) {
    const float* x      = (const float*)d_in[0];
    const int*   ei     = (const int*)  d_in[1];
    // d_in[2] = batch (unused)
    const float* W1     = (const float*)d_in[3];
    const float* a1_src = (const float*)d_in[4];
    const float* a1_dst = (const float*)d_in[5];
    const float* b1     = (const float*)d_in[6];
    const float* W2     = (const float*)d_in[7];
    const float* a2_src = (const float*)d_in[8];
    const float* a2_dst = (const float*)d_in[9];
    const float* b2     = (const float*)d_in[10];
    float* out = (float*)d_out;

    float* w    = (float*)d_ws;
    float* h1   = w + OFF_H1;     // also reused as z after activation
    float* out1 = w + OFF_OUT1;
    float* a1s  = w + OFF_A1S;
    float* a1d  = w + OFF_A1D;
    float* m1   = w + OFF_M1;
    float* s1   = w + OFF_S1;
    float* h2   = w + OFF_H2;
    float* a2s  = w + OFF_A2S;
    float* a2d  = w + OFF_A2D;
    float* m2   = w + OFF_M2;
    float* s2   = w + OFF_S2;
    float* out2 = w + OFF_OUT2;
    _Float16* xh  = (_Float16*)(w + OFF_XH);
    _Float16* w1t = (_Float16*)(w + OFF_W1T);

    const unsigned B = 256;
    const float NEG_INF = -__builtin_huge_valf();

    // init accumulators + operand pre-conversion
    k_fill<<<cdivu(N_NODES*C1, B),    B, 0, stream>>>(out1, N_NODES*C1, 0.f);
    k_fill<<<cdivu(N_NODES*HEADS, B), B, 0, stream>>>(s1,   N_NODES*HEADS, 0.f);
    k_fill<<<cdivu(N_NODES*HEADS, B), B, 0, stream>>>(m1,   N_NODES*HEADS, NEG_INF);
    k_fill<<<cdivu(N_NODES, B),       B, 0, stream>>>(s2,   N_NODES, 0.f);
    k_fill<<<cdivu(N_NODES, B),       B, 0, stream>>>(m2,   N_NODES, NEG_INF);
    k_fill<<<cdivu(N_NODES*N_CLS, B), B, 0, stream>>>(out2, N_NODES*N_CLS, 0.f);
    k_cvt_x<<<cdivu(N_NODES*F_IN/4, B), B, 0, stream>>>(x, xh);
    k_cvt_w1t<<<cdivu(F_IN*C1, B),      B, 0, stream>>>(W1, w1t);

    // layer 1
    const unsigned tiles = (N_NODES / 16) * (C1 / 16);      // 50000 wave-tiles
    k_gemm1_wmma<<<cdivu(tiles, 4), 128, 0, stream>>>(xh, w1t, h1);
    k_alpha1<<<cdivu(N_NODES*HEADS, B), B, 0, stream>>>(h1, a1_src, a1_dst, a1s, a1d);
    k_edge_max1<<<cdivu(E_TOT, B), B, 0, stream>>>(ei, a1s, a1d, m1);
    k_edge_sum1<<<cdivu(E_TOT, B), B, 0, stream>>>(ei, a1s, a1d, m1, s1);
    k_edge_msg1<<<cdivu(E_TOT*HEADS, B), B, 0, stream>>>(ei, a1s, a1d, m1, s1, h1, out1);
    k_act1<<<cdivu(N_NODES*C1, B), B, 0, stream>>>(out1, b1, h1);   // z -> h1 region

    // layer 2
    k_feat2<<<cdivu(N_NODES, B), B, 0, stream>>>(h1, W2, a2_src, a2_dst, h2, a2s, a2d);
    k_edge_max2<<<cdivu(E_TOT, B), B, 0, stream>>>(ei, a2s, a2d, m2);
    k_edge_sum2<<<cdivu(E_TOT, B), B, 0, stream>>>(ei, a2s, a2d, m2, s2);
    k_edge_msg2<<<cdivu(E_TOT, B), B, 0, stream>>>(ei, a2s, a2d, m2, s2, h2, out2);
    k_final<<<cdivu(N_NODES*N_CLS, B), B, 0, stream>>>(out2, b2, out);
}
